// ApproachNet_view_fps_23682449670880
// MI455X (gfx1250) — compile-verified, matrix operand order checked by reading, not run on hardware
//
#include <hip/hip_runtime.h>
#include <hip/hip_bf16.h>

// ---------------- problem constants (match reference) ----------------
#define BB   4
#define NN   20000
#define CC   256
#define VV   300
#define NSMP 1024
#define EPSB 1e-5f
#define GTH  0.09f
#define TWO_PI 6.28318530717958647692f

typedef __attribute__((ext_vector_type(16))) __bf16 v16bf;
typedef __attribute__((ext_vector_type(8)))  float  v8f;
typedef __attribute__((ext_vector_type(8)))  unsigned short us8;  // 16 B
typedef __attribute__((ext_vector_type(4)))  unsigned short us4;  // 8 B

union BfVec { unsigned short u[16]; us8 h[2]; v16bf v; };

__device__ __forceinline__ unsigned short f2bf(float f) {
  unsigned u = __float_as_uint(f);
  unsigned r = u + 0x7FFFu + ((u >> 16) & 1u);   // round-to-nearest-even
  return (unsigned short)(r >> 16);
}

// ---------------- kernel 0: fold BN into weights, convert to bf16 ----------------
__global__ void prep_weights(
    const float* ghw1, const float* ghb1, const float* ghg1, const float* ghbe1,
    const float* ghm1, const float* ghv1,
    const float* w1, const float* b1, const float* g1, const float* be1,
    const float* m1, const float* v1,
    const float* w2, const float* b2, const float* g2, const float* be2,
    const float* m2, const float* v2,
    const float* w3, const float* b3,
    unsigned short* wghp, float* bghp,
    unsigned short* w1p, float* b1p,
    unsigned short* w2p, float* b2p,
    unsigned short* w3p, float* b3p)
{
  int t = blockIdx.x * 256 + threadIdx.x;
  if (t < 65536) { // gh_w1 [256x256]
    int r = t >> 8;
    float s = ghg1[r] * rsqrtf(ghv1[r] + EPSB);
    wghp[t] = f2bf(ghw1[t] * s);
    return;
  }
  t -= 65536;
  if (t < 65536) { // w1 [256x256]
    int r = t >> 8;
    float s = g1[r] * rsqrtf(v1[r] + EPSB);
    w1p[t] = f2bf(w1[t] * s);
    return;
  }
  t -= 65536;
  if (t < 304 * 256) { // w2 [300x256] -> [304x256] zero-padded rows
    int r = t >> 8;
    if (r < VV) {
      float s = g2[r] * rsqrtf(v2[r] + EPSB);
      w2p[t] = f2bf(w2[t] * s);
    } else w2p[t] = 0;
    return;
  }
  t -= 304 * 256;
  if (t < 304 * 320) { // w3 [300x300] -> [304x320] zero-padded rows+cols
    int r = t / 320, c = t % 320;
    unsigned short o = 0;
    if (r < VV && c < VV) o = f2bf(w3[r * VV + c]);
    w3p[t] = o;
    return;
  }
  t -= 304 * 320;
  if (t < 256) { bghp[t] = (ghb1[t] - ghm1[t]) * (ghg1[t] * rsqrtf(ghv1[t] + EPSB)) + ghbe1[t]; return; }
  t -= 256;
  if (t < 256) { b1p[t] = (b1[t] - m1[t]) * (g1[t] * rsqrtf(v1[t] + EPSB)) + be1[t]; return; }
  t -= 256;
  if (t < 304) { b2p[t] = (t < VV) ? (b2[t] - m2[t]) * (g2[t] * rsqrtf(v2[t] + EPSB)) + be2[t] : 0.f; return; }
  t -= 304;
  if (t < 304) { b3p[t] = (t < VV) ? b3[t] : 0.f; return; }
}

// ---------------- kernel 1: stage-1 GEMM fused with graspness reduction ----------
// hidden = relu(W' @ F + b')  [256 x 16cols]  (never stored)
// grasp[col] = sum_row gh_w2[row] * hidden[row,col] + gh_b2
// LDS B tile swizzled [n][K] (stride 264 halves) so each lane's fragment is two
// contiguous 8-half runs -> 2x ds_load_b128 per WMMA.
#define BT_STRIDE 264
__global__ __launch_bounds__(512)
void graspness_kernel(const float* __restrict__ feat,
                      const unsigned short* __restrict__ wgh,
                      const float* __restrict__ bgh,
                      const float* __restrict__ ghw2,
                      const float* __restrict__ ghb2,
                      float* __restrict__ grasp_out)
{
  int b = blockIdx.y, ct = blockIdx.x;
  int tid = threadIdx.x, lane = tid & 31, wave = tid >> 5;
  int idx = lane & 15, sel = lane >> 4;

  __shared__ __align__(16) unsigned short Bt[16 * BT_STRIDE]; // ~8.25 KB
  __shared__ float g16[16];
  if (tid < 16) g16[tid] = 0.f;

  const float* fb = feat + (size_t)b * CC * NN + ct * 16;
  {
    int n = tid & 15, kg = tid >> 4;          // kg in [0,32): 8 consecutive K each
    us8 pack;
#pragma unroll
    for (int j = 0; j < 8; ++j)
      pack[j] = f2bf(fb[(size_t)(kg * 8 + j) * NN + n]);
    *(us8*)&Bt[n * BT_STRIDE + kg * 8] = pack;  // ds_store_b128
  }
  __syncthreads();

  int m0 = wave * 16;                    // wave-uniform -> EXEC stays full
  const unsigned short* arow = wgh + (size_t)(m0 + idx) * CC + sel * 8;
  const unsigned short* brow = &Bt[idx * BT_STRIDE + sel * 8];
  v8f acc = {};
#pragma unroll
  for (int kb = 0; kb < 8; ++kb) {
    BfVec a, bbm;
    a.h[0]   = *(const us8*)(arow + kb * 32);        // K run [8*sel, +8)
    a.h[1]   = *(const us8*)(arow + kb * 32 + 16);   // K run [16+8*sel, +8)
    bbm.h[0] = *(const us8*)(brow + kb * 32);        // ds_load_b128
    bbm.h[1] = *(const us8*)(brow + kb * 32 + 16);   // ds_load_b128
    acc = __builtin_amdgcn_wmma_f32_16x16x32_bf16(false, a.v, false, bbm.v,
                                                  (short)0, acc, false, false);
  }

  float part = 0.f;
#pragma unroll
  for (int r = 0; r < 8; ++r) {          // C/D layout: row = m0 + r + 8*sel
    int row = m0 + r + 8 * sel;
    float h = fmaxf(acc[r] + bgh[row], 0.f);
    part += ghw2[row] * h;
  }
  atomicAdd(&g16[idx], part);            // ds_add_f32
  __syncthreads();
  if (tid < 16)
    grasp_out[(size_t)b * NN + ct * 16 + tid] = g16[tid] + ghb2[0];
}

// ---------------- kernel 2: masked furthest point sampling --------------------
#define PPT 20
__global__ __launch_bounds__(1024)
void fps_kernel(const float* __restrict__ xyz, const float* __restrict__ grasp,
                int* __restrict__ inds_i, float* __restrict__ inds_f)
{
  int b = blockIdx.x, t = threadIdx.x, lane = t & 31, wave = t >> 5;
  __shared__ float bval[32];
  __shared__ int   bidx[32];
  __shared__ int   lastSh;

  const float* xb = xyz + (size_t)b * NN * 3;
  const float* gb = grasp + (size_t)b * NN;

  float px[PPT], py[PPT], pz[PPT], mind[PPT];
  int gidx[PPT]; bool mv[PPT];
#pragma unroll
  for (int j = 0; j < PPT; ++j) {
    int p = t + j * 1024;
    bool in = p < NN;
    gidx[j] = p;
    px[j] = in ? xb[p * 3 + 0] : 0.f;
    py[j] = in ? xb[p * 3 + 1] : 0.f;
    pz[j] = in ? xb[p * 3 + 2] : 0.f;
    mv[j] = in && (gb[in ? p : 0] > GTH);
    mind[j] = mv[j] ? 1e10f : -1.0f;
  }

  // first valid index (argmax of mask == first True; 0 if none)
  int fm = 0x7fffffff;
#pragma unroll
  for (int j = 0; j < PPT; ++j) if (mv[j]) fm = min(fm, gidx[j]);
  for (int off = 16; off; off >>= 1) fm = min(fm, __shfl_xor(fm, off, 32));
  if (lane == 0) bidx[wave] = fm;
  __syncthreads();
  if (wave == 0) {
    int v = bidx[lane];
    for (int off = 16; off; off >>= 1) v = min(v, __shfl_xor(v, off, 32));
    if (lane == 0) lastSh = (v == 0x7fffffff) ? 0 : v;
  }
  __syncthreads();

  for (int s = 0; s < NSMP; ++s) {
    int last = lastSh;
    if (t == 0) { inds_i[b * NSMP + s] = last; inds_f[b * NSMP + s] = (float)last; }
    float lx = xb[last * 3 + 0], ly = xb[last * 3 + 1], lz = xb[last * 3 + 2];

    float bv = -2.f; int bi = 0x7fffffff;
#pragma unroll
    for (int j = 0; j < PPT; ++j) {
      float dx = px[j] - lx, dy = py[j] - ly, dz = pz[j] - lz;
      float d = dx * dx + dy * dy + dz * dz;
      float cand = mv[j] ? d : -1.f;
      mind[j] = fminf(mind[j], cand);
      if (mind[j] > bv || (mind[j] == bv && gidx[j] < bi)) { bv = mind[j]; bi = gidx[j]; }
    }
    for (int off = 16; off; off >>= 1) {
      float ov = __shfl_xor(bv, off, 32);
      int   oi = __shfl_xor(bi, off, 32);
      if (ov > bv || (ov == bv && oi < bi)) { bv = ov; bi = oi; }
    }
    if (lane == 0) { bval[wave] = bv; bidx[wave] = bi; }
    __syncthreads();
    if (wave == 0) {
      float v2 = bval[lane]; int i2 = bidx[lane];
      for (int off = 16; off; off >>= 1) {
        float ov = __shfl_xor(v2, off, 32);
        int   oi = __shfl_xor(i2, off, 32);
        if (ov > v2 || (ov == v2 && oi < i2)) { v2 = ov; i2 = oi; }
      }
      if (lane == 0) lastSh = i2;
    }
    __syncthreads();
  }
}

// ---------------- kernel 3: gathers --------------------------------------------
__global__ void gather_feat(const float* __restrict__ feat,
                            const int* __restrict__ inds,
                            float* __restrict__ outF)
{
  size_t t = (size_t)blockIdx.x * 256 + threadIdx.x;  // [b][c][s]
  if (t >= (size_t)BB * CC * NSMP) return;
  int s = (int)(t & (NSMP - 1));
  int c = (int)((t >> 10) & (CC - 1));
  int b = (int)(t >> 18);
  int ind = inds[b * NSMP + s];
  outF[t] = feat[(size_t)b * CC * NN + (size_t)c * NN + ind];
}

__global__ void gather_small(const float* __restrict__ xyz,
                             const float* __restrict__ grasp,
                             const int* __restrict__ inds,
                             float* __restrict__ oxyz, float* __restrict__ og)
{
  int t = blockIdx.x * 256 + threadIdx.x;
  if (t >= BB * NSMP) return;
  int b = t >> 10;
  int ind = inds[t];
  oxyz[t * 3 + 0] = xyz[(size_t)b * NN * 3 + ind * 3 + 0];
  oxyz[t * 3 + 1] = xyz[(size_t)b * NN * 3 + ind * 3 + 1];
  oxyz[t * 3 + 2] = xyz[(size_t)b * NN * 3 + ind * 3 + 2];
  og[t] = grasp[(size_t)b * NN + ind];
}

// ---------------- kernel 4: generic WMMA GEMM + bias + optional ReLU -----------
// Out[b][row][col] = act(sum_k A[row][k]*B[b][k][col] + bias[row]),
// A bf16 [Mpad x lda] row-major, B f32, Ncols = 1024 fixed.
// Per-slab LDS tile swizzled [n][32] with stride 40 halves.
#define GT_STRIDE 40
__global__ __launch_bounds__(128)
void gemm_bias_act(const unsigned short* __restrict__ A,
                   const float* __restrict__ bias,
                   const float* __restrict__ Bsrc,
                   float* __restrict__ Out,
                   int M, int Mpad, int K, int lda, int ldb,
                   long strideB, long strideO, int relu)
{
  int b = blockIdx.z, ct = blockIdx.x;
  int tid = threadIdx.x, lane = tid & 31, wave = tid >> 5;
  int idx = lane & 15, sel = lane >> 4;
  int m0 = blockIdx.y * 64 + wave * 16;   // wave-uniform
  bool act = (m0 < Mpad);

  __shared__ __align__(16) unsigned short Bt[16 * GT_STRIDE];
  const float* bp = Bsrc + (size_t)b * strideB + ct * 16;
  int sn = tid & 15, skg = tid >> 4;      // staging: col sn, K group skg (4 K each)

  const unsigned short* arow = A + (size_t)(m0 + idx) * lda + sel * 8;
  const unsigned short* brow = &Bt[idx * GT_STRIDE + sel * 8];

  v8f acc = {};
  int nkb = lda >> 5;
  for (int kb = 0; kb < nkb; ++kb) {
    __syncthreads();
    {
      us4 pack;
#pragma unroll
      for (int j = 0; j < 4; ++j) {
        int gk = kb * 32 + skg * 4 + j;
        pack[j] = (gk < K) ? f2bf(bp[(size_t)gk * ldb + sn]) : (unsigned short)0;
      }
      *(us4*)&Bt[sn * GT_STRIDE + skg * 4] = pack;   // ds_store_b64
    }
    if (kb + 1 < nkb && ((kb * 32 + 32) < K))
      __builtin_prefetch(bp + (size_t)(kb * 32 + 32) * ldb, 0, 3);
    __syncthreads();
    if (act) {
      BfVec a, bbm;
      a.h[0]   = *(const us8*)(arow + kb * 32);       // global_load_b128
      a.h[1]   = *(const us8*)(arow + kb * 32 + 16);
      bbm.h[0] = *(const us8*)(brow);                 // ds_load_b128
      bbm.h[1] = *(const us8*)(brow + 16);
      acc = __builtin_amdgcn_wmma_f32_16x16x32_bf16(false, a.v, false, bbm.v,
                                                    (short)0, acc, false, false);
    }
  }
  if (act) {
#pragma unroll
    for (int r = 0; r < 8; ++r) {
      int row = m0 + r + 8 * sel;
      if (row < M) {
        float vv = acc[r] + bias[row];
        if (relu) vv = fmaxf(vv, 0.f);
        Out[(size_t)b * strideO + (size_t)row * NSMP + ct * 16 + idx] = vv;
      }
    }
  }
}

// ---------------- kernel 5: view_score transpose, max/argmax, views, rotations --
__global__ void view_post(const float* __restrict__ f3,
                          float* __restrict__ vs_out, float* __restrict__ ti_out,
                          float* __restrict__ ts_out, float* __restrict__ vpxyz,
                          float* __restrict__ vprot)
{
  int t = blockIdx.x * 256 + threadIdx.x;
  if (t >= BB * NSMP) return;
  int s = t & (NSMP - 1);
  int b = t >> 10;
  const float* fb = f3 + (size_t)b * VV * NSMP + s;

  float bv = -1e30f; int bi = 0;
  for (int v = 0; v < VV; ++v) {
    float x = fb[(size_t)v * NSMP];
    vs_out[(size_t)b * NSMP * VV + (size_t)s * VV + v] = x;
    if (x > bv) { bv = x; bi = v; }
  }
  ts_out[t] = bv;
  ti_out[t] = (float)bi;

  // fibonacci sphere view
  const float PHI = 0.61803398874989484820f;
  float z = (2.f * bi + 1.f) / (float)VV - 1.f;
  float r = sqrtf(fmaxf(1.f - z * z, 0.f));
  float ang = TWO_PI * (float)bi * PHI;
  float vx = r * cosf(ang), vy = r * sinf(ang), vz = z;
  vpxyz[t * 3 + 0] = vx; vpxyz[t * 3 + 1] = vy; vpxyz[t * 3 + 2] = vz;

  // viewpoint_to_matrix(-view, angle=0) -> R = [axis_x | axis_y | axis_z]
  float tx = -vx, ty = -vy, tz = -vz;
  float ayx = -ty, ayy = tx, ayz = 0.f;
  float ny = sqrtf(ayx * ayx + ayy * ayy);
  if (ny == 0.f) { ayx = 0.f; ayy = 1.f; ayz = 0.f; }
  float nx = sqrtf(tx * tx + ty * ty + tz * tz);
  nx = (nx == 0.f) ? 1.f : nx;
  float axx = tx / nx, axy = ty / nx, axz = tz / nx;
  float ny2 = sqrtf(ayx * ayx + ayy * ayy + ayz * ayz);
  ny2 = (ny2 == 0.f) ? 1.f : ny2;
  ayx /= ny2; ayy /= ny2; ayz /= ny2;
  float azx = axy * ayz - axz * ayy;
  float azy = axz * ayx - axx * ayz;
  float azz = axx * ayy - axy * ayx;
  float* R = vprot + (size_t)t * 9;      // row-major 3x3, columns = axes
  R[0] = axx; R[1] = ayx; R[2] = azx;
  R[3] = axy; R[4] = ayy; R[5] = azy;
  R[6] = axz; R[7] = ayz; R[8] = azz;
}

// =================================================================================
extern "C" void kernel_launch(void* const* d_in, const int* in_sizes, int n_in,
                              void* d_out, int out_size, void* d_ws, size_t ws_size,
                              hipStream_t stream) {
  const float* seed_xyz  = (const float*)d_in[0];
  const float* seed_feat = (const float*)d_in[1];
  const float* gh_w1 = (const float*)d_in[2];
  const float* gh_b1 = (const float*)d_in[3];
  const float* gh_g1 = (const float*)d_in[4];
  const float* gh_be1 = (const float*)d_in[5];
  const float* gh_m1 = (const float*)d_in[6];
  const float* gh_v1 = (const float*)d_in[7];
  const float* gh_w2 = (const float*)d_in[8];
  const float* gh_b2 = (const float*)d_in[9];
  const float* w1 = (const float*)d_in[10];
  const float* b1 = (const float*)d_in[11];
  const float* g1 = (const float*)d_in[12];
  const float* be1 = (const float*)d_in[13];
  const float* m1 = (const float*)d_in[14];
  const float* v1 = (const float*)d_in[15];
  const float* w2 = (const float*)d_in[16];
  const float* b2 = (const float*)d_in[17];
  const float* g2 = (const float*)d_in[18];
  const float* be2 = (const float*)d_in[19];
  const float* m2 = (const float*)d_in[20];
  const float* v2 = (const float*)d_in[21];
  const float* w3 = (const float*)d_in[22];
  const float* b3 = (const float*)d_in[23];

  float* out = (float*)d_out;
  // output offsets (floats), reference tuple order
  const size_t o_grasp = 0;                          // [B,N]
  const size_t o_inds  = o_grasp + (size_t)BB * NN;  // [B,NS]
  const size_t o_xyz   = o_inds + (size_t)BB * NSMP; // [B,NS,3]
  const size_t o_feat  = o_xyz + (size_t)BB * NSMP * 3;      // [B,C,NS]
  const size_t o_fp2g  = o_feat + (size_t)BB * CC * NSMP;    // [B,NS]
  const size_t o_vs    = o_fp2g + (size_t)BB * NSMP;         // [B,NS,V]
  const size_t o_tvi   = o_vs + (size_t)BB * NSMP * VV;      // [B,NS]
  const size_t o_tvs   = o_tvi + (size_t)BB * NSMP;          // [B,NS]
  const size_t o_vpx   = o_tvs + (size_t)BB * NSMP;          // [B,NS,3]
  const size_t o_rot   = o_vpx + (size_t)BB * NSMP * 3;      // [B,NS,3,3]

  // workspace layout
  char* wsb = (char*)d_ws;
  size_t off = 0;
  auto alloc = [&](size_t bytes) { size_t o = off; off += (bytes + 255) & ~(size_t)255; return o; };
  unsigned short* wghp = (unsigned short*)(wsb + alloc(65536 * 2));
  float*          bghp = (float*)(wsb + alloc(256 * 4));
  unsigned short* w1p  = (unsigned short*)(wsb + alloc(65536 * 2));
  float*          b1p  = (float*)(wsb + alloc(256 * 4));
  unsigned short* w2p  = (unsigned short*)(wsb + alloc(304 * 256 * 2));
  float*          b2p  = (float*)(wsb + alloc(304 * 4));
  unsigned short* w3p  = (unsigned short*)(wsb + alloc(304 * 320 * 2));
  float*          b3p  = (float*)(wsb + alloc(304 * 4));
  int*            inds = (int*)(wsb + alloc((size_t)BB * NSMP * 4));
  float*          f1   = (float*)(wsb + alloc((size_t)BB * 256 * NSMP * 4));
  float*          f2   = (float*)(wsb + alloc((size_t)BB * 304 * NSMP * 4));
  float*          f3   = (float*)(wsb + alloc((size_t)BB * VV * NSMP * 4));
  (void)ws_size;

  // 0) weight prep (BN-folded bf16)
  {
    int total = 65536 + 65536 + 304 * 256 + 304 * 320 + 256 + 256 + 304 + 304;
    prep_weights<<<(total + 255) / 256, 256, 0, stream>>>(
        gh_w1, gh_b1, gh_g1, gh_be1, gh_m1, gh_v1,
        w1, b1, g1, be1, m1, v1, w2, b2, g2, be2, m2, v2, w3, b3,
        wghp, bghp, w1p, b1p, w2p, b2p, w3p, b3p);
  }
  // 1) stage-1 GEMM + fused graspness
  graspness_kernel<<<dim3(NN / 16, BB), 512, 0, stream>>>(
      seed_feat, wghp, bghp, gh_w2, gh_b2, out + o_grasp);
  // 2) masked FPS
  fps_kernel<<<BB, 1024, 0, stream>>>(seed_xyz, out + o_grasp, inds, out + o_inds);
  // 3) gathers
  gather_feat<<<(BB * CC * NSMP) / 256, 256, 0, stream>>>(seed_feat, inds, out + o_feat);
  gather_small<<<(BB * NSMP + 255) / 256, 256, 0, stream>>>(
      seed_xyz, out + o_grasp, inds, out + o_xyz, out + o_fp2g);
  // 4) stage-2 GEMMs (WMMA bf16, f32 accumulate)
  gemm_bias_act<<<dim3(NSMP / 16, 4, BB), 128, 0, stream>>>(
      w1p, b1p, out + o_feat, f1, 256, 256, 256, 256, NSMP,
      (long)CC * NSMP, (long)256 * NSMP, 1);
  gemm_bias_act<<<dim3(NSMP / 16, 5, BB), 128, 0, stream>>>(
      w2p, b2p, f1, f2, VV, 304, 256, 256, NSMP,
      (long)256 * NSMP, (long)304 * NSMP, 1);
  gemm_bias_act<<<dim3(NSMP / 16, 5, BB), 128, 0, stream>>>(
      w3p, b3p, f2, f3, VV, 304, VV, 320, NSMP,
      (long)304 * NSMP, (long)VV * NSMP, 0);
  // 5) view post-processing
  view_post<<<(BB * NSMP + 255) / 256, 256, 0, stream>>>(
      f3, out + o_vs, out + o_tvi, out + o_tvs, out + o_vpx, out + o_rot);
}
